// MultiScaleTrajectoryEncoder_82497731821916
// MI455X (gfx1250) — compile-verified, matrix operand order checked by reading, not run on hardware
//
#include <hip/hip_runtime.h>
#include <hip/hip_fp16.h>

// ---- CDNA5 WMMA types ----
typedef __attribute__((ext_vector_type(16))) _Float16 v16h;
typedef __attribute__((ext_vector_type(8)))  float    v8f;

#define WMMA_F16(a,b,c) __builtin_amdgcn_wmma_f32_16x16x32_f16(false,(a),false,(b),(short)0,(c),false,false)

__device__ __forceinline__ v8f zero8(){
  v8f z;
  #pragma unroll
  for(int i=0;i<8;++i) z[i]=0.f;
  return z;
}
__device__ __forceinline__ v16h zero16(){
  v16h z;
  #pragma unroll
  for(int i=0;i<16;++i) z[i]=(_Float16)0.f;
  return z;
}

// K index striping for 16-bit A/B operands (16x32 tile):
// lanes 0-15 hold K {0..7,16..23}, lanes 16-31 hold K {8..15,24..31}
__device__ __forceinline__ int a_k(int i, int h16){ return (i&7) + ((i>>3)<<4) + (h16<<3); }

// tile element (lane%16 [row], k) = src[(lane%16)*ld + k]   (f32 source -> f16 regs)
__device__ __forceinline__ v16h load_tile_f32(const float* __restrict__ src, int ld, int lane){
  const float* rb = src + (size_t)(lane&15)*ld;
  int h16=(lane>>4)&1; v16h r;
  #pragma unroll
  for(int i=0;i<16;++i) r[i]=(_Float16)rb[a_k(i,h16)];
  return r;
}
__device__ __forceinline__ v16h load_tile_f16(const _Float16* __restrict__ src, int ld, int lane){
  const _Float16* rb = src + (size_t)(lane&15)*ld;
  int h16=(lane>>4)&1; v16h r;
  #pragma unroll
  for(int i=0;i<16;++i) r[i]=rb[a_k(i,h16)];
  return r;
}
// B tile from K-major storage: element (k, lane%16) = src[k*ld + lane%16]
__device__ __forceinline__ v16h load_tile_kmajor_f16(const _Float16* __restrict__ src, int ld, int lane){
  const _Float16* cb = src + (lane&15);
  int h16=(lane>>4)&1; v16h r;
  #pragma unroll
  for(int i=0;i<16;++i) r[i]=cb[(size_t)a_k(i,h16)*ld];
  return r;
}
// Pre-swizzled tile: one contiguous 32B per lane (two b128 loads)
__device__ __forceinline__ v16h load_tile_sw(const _Float16* src, int tile, int lane){
  return *reinterpret_cast<const v16h*>(src + (((size_t)tile*32 + lane) << 4));
}

__device__ __forceinline__ float rmax16(float x){
  #pragma unroll
  for(int m=8;m;m>>=1) x=fmaxf(x,__shfl_xor(x,m,32));
  return x;
}
__device__ __forceinline__ float rsum16(float x){
  #pragma unroll
  for(int m=8;m;m>>=1) x+=__shfl_xor(x,m,32);
  return x;
}
__device__ __forceinline__ float sigf(float x){ return 1.f/(1.f+__expf(-x)); }

// dims
#define PB  8     // patch length
#define DD  128   // input dim
#define HH  64    // hidden
#define SQ  512   // transformer sequence

// ===== weight swizzle: W(rows=out, cols=K) f32 -> per-tile WMMA B layout f16 =====
__global__ __launch_bounds__(256) void k_swz(const float* __restrict__ W,
                                             _Float16* __restrict__ out, int K){
  int idx = blockIdx.x*256 + threadIdx.x;
  int i    = idx & 15;
  int lane = (idx >> 4) & 31;
  int tile = idx >> 9;
  int ktiles = K >> 5;
  int kt = tile % ktiles, jt = tile / ktiles;
  out[idx] = (_Float16)W[(size_t)(jt*16 + (lane&15))*K + kt*32 + a_k(i,(lane>>4)&1)];
}

// ======================= GRU + in-patch mean pool =======================
// wave = 16 patches; gates fused per column-group j. WhhT staged in LDS;
// Wih tile loads kept in-loop (L0 hits) via a laundered zero offset that
// preserves global-addrspace provenance (global_load, not flat_load).
__global__ __launch_bounds__(256) void k_gru(
    const float* __restrict__ x, const _Float16* __restrict__ WihT,
    const _Float16* __restrict__ WhhT, const float* __restrict__ bih,
    const float* __restrict__ bhh, float* __restrict__ seq)
{
  __shared__ _Float16 hscr[8][16*64];     // 16 KB: h relayout scratch
  __shared__ _Float16 WhhL[192*64];       // 24 KB: swizzled Whh tiles
  int lane=threadIdx.x&31, wave=threadIdx.x>>5;
  int pbase=(blockIdx.x*8+wave)*16;
  int colb=lane&15, rowoff=(lane>>4)*8;

  { // cooperative stage of WhhT into LDS
    const uint32_t* wsrc=(const uint32_t*)WhhT;
    uint32_t* wdst=(uint32_t*)WhhL;
    for(int i=threadIdx.x;i<192*64/2;i+=256) wdst[i]=wsrc[i];
  }
  __syncthreads();

  float bi[12], bh[12];
  #pragma unroll
  for(int j=0;j<12;++j){ bi[j]=bih[j*16+colb]; bh[j]=bhh[j*16+colb]; }

  v8f hT[4], hs[4];
  #pragma unroll
  for(int j=0;j<4;++j){ hT[j]=zero8(); hs[j]=zero8(); }
  v16h hA[2]; hA[0]=zero16(); hA[1]=zero16();
  _Float16* hl=&hscr[wave][0];

  for(int t=0;t<PB;++t){
    // loop-variant (but always zero) offset: defeats LICM hoist+spill of the
    // 48 Wih tiles while keeping the loads on the global path.
    size_t zoff = 0;
    asm volatile("" : "+s"(zoff));
    const _Float16* Wt = WihT + zoff;
    const float* xsrc = x + ((size_t)pbase*PB + t)*DD;
    if(t+1<PB) __builtin_prefetch(xsrc+DD,0,1);
    v16h xA[4];
    #pragma unroll
    for(int kk=0;kk<4;++kk) xA[kk]=load_tile_f32(xsrc+kk*32, PB*DD, lane);

    #pragma unroll
    for(int j=0;j<4;++j){
      // r gate (cols 16j..)
      v8f gr=zero8();
      #pragma unroll
      for(int kk=0;kk<4;++kk)
        gr=WMMA_F16(xA[kk], load_tile_sw(Wt, j*4+kk, lane), gr);
      #pragma unroll
      for(int kk=0;kk<2;++kk)
        gr=WMMA_F16(hA[kk], load_tile_sw(WhhL, j*2+kk, lane), gr);
      // z gate (cols 64+16j..)
      v8f gz=zero8();
      #pragma unroll
      for(int kk=0;kk<4;++kk)
        gz=WMMA_F16(xA[kk], load_tile_sw(Wt, (4+j)*4+kk, lane), gz);
      #pragma unroll
      for(int kk=0;kk<2;++kk)
        gz=WMMA_F16(hA[kk], load_tile_sw(WhhL, (4+j)*2+kk, lane), gz);
      // n gate (cols 128+16j..), x and h parts kept separate
      v8f gnx=zero8();
      #pragma unroll
      for(int kk=0;kk<4;++kk)
        gnx=WMMA_F16(xA[kk], load_tile_sw(Wt, (8+j)*4+kk, lane), gnx);
      v8f gnh=zero8();
      #pragma unroll
      for(int kk=0;kk<2;++kk)
        gnh=WMMA_F16(hA[kk], load_tile_sw(WhhL, (8+j)*2+kk, lane), gnh);
      #pragma unroll
      for(int e=0;e<8;++e){
        float r =sigf(gr[e]+bi[j]+bh[j]);
        float z =sigf(gz[e]+bi[4+j]+bh[4+j]);
        float nv=tanhf(gnx[e]+bi[8+j]+r*(gnh[e]+bh[8+j]));
        float hn=(1.f-z)*nv + z*hT[j][e];
        hT[j][e]=hn; hs[j][e]+=hn;
      }
    }
    // h: C layout -> LDS -> A layout (f16) for next step
    #pragma unroll
    for(int j=0;j<4;++j){
      #pragma unroll
      for(int v=0;v<8;++v)
        hl[(v+rowoff)*64 + j*16 + colb]=(_Float16)hT[j][v];
    }
    #pragma unroll
    for(int kk=0;kk<2;++kk) hA[kk]=load_tile_f16(hl+kk*32, 64, lane);
  }
  float* sb=seq+(size_t)pbase*HH;
  #pragma unroll
  for(int j=0;j<4;++j){
    #pragma unroll
    for(int v=0;v<8;++v)
      sb[(size_t)(v+rowoff)*HH + j*16 + colb]=hs[j][v]*0.125f;
  }
}

// ======================= QKV projection =======================
// Writes Q,K padded to hd=32 (zeros, Q pre-scaled 0.25) and V (hd=16), all f16.
__global__ __launch_bounds__(256) void k_qkv(
    const float* __restrict__ X, const _Float16* __restrict__ WqkvT,
    const float* __restrict__ bqkv, _Float16* __restrict__ Qp,
    _Float16* __restrict__ Kp, _Float16* __restrict__ Vh)
{
  int lane=threadIdx.x&31, wave=threadIdx.x>>5;
  size_t rbase=((size_t)blockIdx.x*8+wave)*16;
  int b=(int)(rbase>>9), s=(int)(rbase&511);
  int colb=lane&15, rowoff=(lane>>4)*8;
  v16h A[2];
  #pragma unroll
  for(int kk=0;kk<2;++kk) A[kk]=load_tile_f32(X+rbase*HH+kk*32, HH, lane);

  #pragma unroll
  for(int j=0;j<12;++j){
    v8f acc=zero8();
    #pragma unroll
    for(int kk=0;kk<2;++kk)
      acc=WMMA_F16(A[kk], load_tile_sw(WqkvT, j*2+kk, lane), acc);
    float bias=bqkv[j*16+colb];
    int h=j&3;
    size_t bh=(size_t)b*4+h;
    if(j<4){
      _Float16* dst=Qp+(bh*SQ+s)*32;
      #pragma unroll
      for(int v=0;v<8;++v){
        size_t ro=(size_t)(v+rowoff)*32;
        dst[ro+colb]=(_Float16)((acc[v]+bias)*0.25f);
        dst[ro+16+colb]=(_Float16)0.f;
      }
    } else if(j<8){
      _Float16* dst=Kp+(bh*SQ+s)*32;
      #pragma unroll
      for(int v=0;v<8;++v){
        size_t ro=(size_t)(v+rowoff)*32;
        dst[ro+colb]=(_Float16)(acc[v]+bias);
        dst[ro+16+colb]=(_Float16)0.f;
      }
    } else {
      _Float16* dst=Vh+(bh*SQ+s)*16;
      #pragma unroll
      for(int v=0;v<8;++v)
        dst[(size_t)(v+rowoff)*16+colb]=(_Float16)(acc[v]+bias);
    }
  }
}

// ======================= Flash attention (per batch*head WG) =======================
__global__ __launch_bounds__(256) void k_attn(
    const _Float16* __restrict__ Qp, const _Float16* __restrict__ Kp,
    const _Float16* __restrict__ Vh, float* __restrict__ O)
{
  __shared__ _Float16 Ks[SQ*32];
  __shared__ _Float16 Vs[SQ*16];
  __shared__ _Float16 Ps[8][16*32];
  int bh=blockIdx.x, b=bh>>2, h=bh&3;
  int tid=threadIdx.x, lane=tid&31, wave=tid>>5;
  int colb=lane&15, rowoff=(lane>>4)*8;

  { // cooperative stage of K,V head-slices into LDS (32KB + 16KB)
    const uint32_t* ksrc=(const uint32_t*)(Kp+(size_t)bh*SQ*32);
    uint32_t* kdst=(uint32_t*)Ks;
    for(int i=tid;i<SQ*16;i+=256) kdst[i]=ksrc[i];
    const uint32_t* vsrc=(const uint32_t*)(Vh+(size_t)bh*SQ*16);
    uint32_t* vdst=(uint32_t*)Vs;
    for(int i=tid;i<SQ*8;i+=256) vdst[i]=vsrc[i];
  }
  __syncthreads();

  _Float16* pl=&Ps[wave][0];
  for(int qt=0;qt<4;++qt){
    int qbase=wave*64+qt*16;
    v16h Qa=load_tile_f16(Qp+((size_t)bh*SQ+qbase)*32, 32, lane);
    v8f acc=zero8(), mrow, lrow;
    #pragma unroll
    for(int e=0;e<8;++e){ mrow[e]=-1e30f; lrow[e]=0.f; }

    for(int kc=0;kc<16;++kc){      // 32 keys / chunk
      int kbase=kc*32;
      v16h Kb0=load_tile_f16(Ks+(size_t)kbase*32, 32, lane);
      v16h Kb1=load_tile_f16(Ks+(size_t)(kbase+16)*32, 32, lane);
      v8f s0=WMMA_F16(Qa,Kb0,zero8());
      v8f s1=WMMA_F16(Qa,Kb1,zero8());
      v8f nmv, scv;
      #pragma unroll
      for(int e=0;e<8;++e){
        float cm=fmaxf(rmax16(s0[e]),rmax16(s1[e]));
        float nm=fmaxf(mrow[e],cm);
        nmv[e]=nm; scv[e]=__expf(mrow[e]-nm); mrow[e]=nm;
      }
      #pragma unroll
      for(int e=0;e<8;++e){
        s0[e]=__expf(s0[e]-nmv[e]);
        s1[e]=__expf(s1[e]-nmv[e]);
        lrow[e]=lrow[e]*scv[e]+rsum16(s0[e])+rsum16(s1[e]);
        acc[e]*=scv[e];
      }
      // P chunk -> LDS (C layout) -> A layout
      #pragma unroll
      for(int e=0;e<8;++e){
        pl[(e+rowoff)*32+colb]=(_Float16)s0[e];
        pl[(e+rowoff)*32+16+colb]=(_Float16)s1[e];
      }
      v16h Pa=load_tile_f16(pl, 32, lane);
      v16h Vb=load_tile_kmajor_f16(Vs+(size_t)kbase*16, 16, lane);
      acc=WMMA_F16(Pa,Vb,acc);
    }
    float* ob=O+((size_t)b*SQ+qbase)*HH + h*16;
    #pragma unroll
    for(int e=0;e<8;++e)
      ob[(size_t)(e+rowoff)*HH+colb]=acc[e]/lrow[e];
  }
}

// ======================= out-proj + residual + LayerNorm =======================
__global__ __launch_bounds__(256) void k_projln(
    float* __restrict__ X, const float* __restrict__ Y,
    const _Float16* __restrict__ WoT, const float* __restrict__ bo,
    const float* __restrict__ g, const float* __restrict__ be)
{
  int lane=threadIdx.x&31, wave=threadIdx.x>>5;
  size_t rbase=((size_t)blockIdx.x*8+wave)*16;
  int colb=lane&15, rowoff=(lane>>4)*8;
  v16h A[2];
  #pragma unroll
  for(int kk=0;kk<2;++kk) A[kk]=load_tile_f32(Y+rbase*HH+kk*32, HH, lane);
  v8f o[4];
  #pragma unroll
  for(int j=0;j<4;++j){
    v8f acc=zero8();
    #pragma unroll
    for(int kk=0;kk<2;++kk)
      acc=WMMA_F16(A[kk], load_tile_sw(WoT, j*2+kk, lane), acc);
    float bias=bo[j*16+colb];
    const float* xr=X+rbase*HH+j*16+colb;
    #pragma unroll
    for(int e=0;e<8;++e) o[j][e]=acc[e]+bias+xr[(size_t)(e+rowoff)*HH];
  }
  float gv[4],bv[4];
  #pragma unroll
  for(int j=0;j<4;++j){ gv[j]=g[j*16+colb]; bv[j]=be[j*16+colb]; }
  float* xw=X+rbase*HH;
  #pragma unroll
  for(int e=0;e<8;++e){
    float s=o[0][e]+o[1][e]+o[2][e]+o[3][e];
    float mu=rsum16(s)*(1.f/64.f);
    float d0=o[0][e]-mu,d1=o[1][e]-mu,d2=o[2][e]-mu,d3=o[3][e]-mu;
    float var=rsum16(d0*d0+d1*d1+d2*d2+d3*d3)*(1.f/64.f);
    float inv=rsqrtf(var+1e-5f);
    #pragma unroll
    for(int j=0;j<4;++j)
      xw[(size_t)(e+rowoff)*HH+j*16+colb]=(o[j][e]-mu)*inv*gv[j]+bv[j];
  }
}

// ======================= FF (relu) + residual + LayerNorm =======================
__global__ __launch_bounds__(128) void k_ffln(
    float* __restrict__ X, const _Float16* __restrict__ W1T,
    const float* __restrict__ b1, const _Float16* __restrict__ W2T,
    const float* __restrict__ b2, const float* __restrict__ g,
    const float* __restrict__ be)
{
  __shared__ _Float16 h1[4][16*256];
  int lane=threadIdx.x&31, wave=threadIdx.x>>5;
  size_t rbase=((size_t)blockIdx.x*4+wave)*16;
  int colb=lane&15, rowoff=(lane>>4)*8;
  v16h A[2];
  #pragma unroll
  for(int kk=0;kk<2;++kk) A[kk]=load_tile_f32(X+rbase*HH+kk*32, HH, lane);
  _Float16* hl=&h1[wave][0];
  #pragma unroll
  for(int j=0;j<16;++j){
    v8f acc=zero8();
    #pragma unroll
    for(int kk=0;kk<2;++kk)
      acc=WMMA_F16(A[kk], load_tile_sw(W1T, j*2+kk, lane), acc);
    float bias=b1[j*16+colb];
    #pragma unroll
    for(int e=0;e<8;++e)
      hl[(size_t)(e+rowoff)*256+j*16+colb]=(_Float16)fmaxf(acc[e]+bias,0.f);
  }
  v16h pA[8];
  #pragma unroll
  for(int kk=0;kk<8;++kk) pA[kk]=load_tile_f16(hl+kk*32, 256, lane);
  v8f o[4];
  #pragma unroll
  for(int j=0;j<4;++j){
    v8f acc=zero8();
    #pragma unroll
    for(int kk=0;kk<8;++kk)
      acc=WMMA_F16(pA[kk], load_tile_sw(W2T, j*8+kk, lane), acc);
    float bias=b2[j*16+colb];
    const float* xr=X+rbase*HH+j*16+colb;
    #pragma unroll
    for(int e=0;e<8;++e) o[j][e]=acc[e]+bias+xr[(size_t)(e+rowoff)*HH];
  }
  float gv[4],bv[4];
  #pragma unroll
  for(int j=0;j<4;++j){ gv[j]=g[j*16+colb]; bv[j]=be[j*16+colb]; }
  float* xw=X+rbase*HH;
  #pragma unroll
  for(int e=0;e<8;++e){
    float s=o[0][e]+o[1][e]+o[2][e]+o[3][e];
    float mu=rsum16(s)*(1.f/64.f);
    float d0=o[0][e]-mu,d1=o[1][e]-mu,d2=o[2][e]-mu,d3=o[3][e]-mu;
    float var=rsum16(d0*d0+d1*d1+d2*d2+d3*d3)*(1.f/64.f);
    float inv=rsqrtf(var+1e-5f);
    #pragma unroll
    for(int j=0;j<4;++j)
      xw[(size_t)(e+rowoff)*HH+j*16+colb]=(o[j][e]-mu)*inv*gv[j]+bv[j];
  }
}

// ======================= final mean over patches =======================
__global__ void k_mean(const float* __restrict__ X, float* __restrict__ out){
  int b=blockIdx.x, j=threadIdx.x;
  float s=0.f;
  for(int np=0;np<SQ;++np) s+=X[((size_t)b*SQ+np)*HH+j];
  out[b*HH+j]=s*(1.f/512.f);
}

extern "C" void kernel_launch(void* const* d_in, const int* in_sizes, int n_in,
                              void* d_out, int out_size, void* d_ws, size_t ws_size,
                              hipStream_t stream){
  const float* x  =(const float*)d_in[0];
  const float* Wih=(const float*)d_in[1];
  const float* Whh=(const float*)d_in[2];
  const float* bih=(const float*)d_in[3];
  const float* bhh=(const float*)d_in[4];

  char* ws=(char*)d_ws;
  float*    X =(float*)ws;                       // 8 MB  (B,512,64) f32
  float*    Y =(float*)(ws+(size_t)(8<<20));     // 8 MB  attention output
  _Float16* Qp=(_Float16*)(ws+(size_t)(16<<20)); // 8 MB  (B,4,512,32) f16
  _Float16* Kp=(_Float16*)(ws+(size_t)(24<<20)); // 8 MB
  _Float16* Vh=(_Float16*)(ws+(size_t)(32<<20)); // 4 MB  (B,4,512,16) f16
  // pre-swizzled f16 weight tiles
  _Float16* WihT=(_Float16*)(ws+(size_t)(36<<20));            // 192*128 halfs
  _Float16* WhhT=(_Float16*)(ws+(size_t)(36<<20)+(64<<10));   // 192*64
  _Float16* LW[2][4];
  for(int L=0;L<2;++L){
    char* base=ws+(size_t)(37<<20)+(size_t)L*(1<<20);
    LW[L][0]=(_Float16*)(base);            // WqkvT 192*64
    LW[L][1]=(_Float16*)(base+(32<<10));   // WoT   64*64
    LW[L][2]=(_Float16*)(base+(64<<10));   // W1T   256*64
    LW[L][3]=(_Float16*)(base+(128<<10));  // W2T   64*256
  }

  // weight prep (cheap, deterministic, graph-safe)
  k_swz<<<96,256,0,stream>>>(Wih, WihT, 128);
  k_swz<<<48,256,0,stream>>>(Whh, WhhT, 64);
  for(int L=0;L<2;++L){
    int base=5+L*12;
    k_swz<<<48,256,0,stream>>>((const float*)d_in[base+0], LW[L][0], 64);
    k_swz<<<16,256,0,stream>>>((const float*)d_in[base+2], LW[L][1], 64);
    k_swz<<<64,256,0,stream>>>((const float*)d_in[base+4], LW[L][2], 64);
    k_swz<<<64,256,0,stream>>>((const float*)d_in[base+6], LW[L][3], 256);
  }

  k_gru<<<256,256,0,stream>>>(x,WihT,WhhT,bih,bhh,X);

  for(int L=0;L<2;++L){
    int base=5+L*12;
    const float* bqkv=(const float*)d_in[base+1];
    const float* bo  =(const float*)d_in[base+3];
    const float* b1  =(const float*)d_in[base+5];
    const float* b2  =(const float*)d_in[base+7];
    const float* g1  =(const float*)d_in[base+8];
    const float* be1 =(const float*)d_in[base+9];
    const float* g2  =(const float*)d_in[base+10];
    const float* be2 =(const float*)d_in[base+11];

    k_qkv   <<<256,256,0,stream>>>(X,LW[L][0],bqkv,Qp,Kp,Vh);
    k_attn  <<<256,256,0,stream>>>(Qp,Kp,Vh,Y);
    k_projln<<<256,256,0,stream>>>(X,Y,LW[L][1],bo,g1,be1);
    k_ffln  <<<512,128,0,stream>>>(X,LW[L][2],b1,LW[L][3],b2,g2,be2);
  }
  k_mean<<<64,64,0,stream>>>(X,(float*)d_out);
}
